// Splatter_70300024701471
// MI455X (gfx1250) — compile-verified, matrix operand order checked by reading, not run on hardware
//
#include <hip/hip_runtime.h>
#include <hip/hip_bf16.h>

typedef float v2f __attribute__((ext_vector_type(2)));
typedef float v8f __attribute__((ext_vector_type(8)));
typedef int   v4i __attribute__((ext_vector_type(4)));
typedef _Float16 v16h __attribute__((ext_vector_type(16)));

#define NPAD 4096
#define NC (NPAD / 16)     // 256 chunks of 16 gaussians
#define CHF 192            // floats per chunk block: 16*8 coeff + 16 opa + 16*3 rgb
#define HH 96
#define WW 96

#define AS1 __attribute__((address_space(1)))
#define AS3 __attribute__((address_space(3)))

#if defined(__has_builtin)
#if __has_builtin(__builtin_amdgcn_wmma_f32_16x16x4_f32)
#define HAVE_WMMA_F32X4 1
#endif
#if __has_builtin(__builtin_amdgcn_global_load_async_to_lds_b128)
#define HAVE_ASYNC 1
#endif
#if __has_builtin(__builtin_amdgcn_s_wait_asynccnt)
#define HAVE_WAIT_ASYNC_BUILTIN 1
#endif
#endif
#ifndef HAVE_WMMA_F32X4
#define HAVE_WMMA_F32X4 0
#endif
#ifndef HAVE_ASYNC
#define HAVE_ASYNC 0
#endif

__device__ __forceinline__ float sigmoidf(float x) {
  return 1.0f / (1.0f + __expf(-x));
}

__device__ __forceinline__ void wait_async() {
#if HAVE_ASYNC
#if defined(HAVE_WAIT_ASYNC_BUILTIN)
  __builtin_amdgcn_s_wait_asynccnt(0);
#else
  asm volatile("s_wait_asynccnt 0x0" ::: "memory");
#endif
#endif
}

// Stage one 768B chunk block (192 floats) global -> LDS.
// Async path: 48 lanes issue one GLOBAL_LOAD_ASYNC_TO_LDS_B128 each (ASYNCcnt).
// Builtin signature (from hipcc diagnostic): arg0 = global int4* (AS1, non-const).
__device__ __forceinline__ void stage_chunk(const float* __restrict__ gsrc,
                                            float* lbuf, int tid) {
#if HAVE_ASYNC
  if (tid < 48) {
    AS1 v4i* g = (AS1 v4i*)(const_cast<float*>(gsrc) + tid * 4);
    AS3 v4i* l = (AS3 v4i*)(lbuf + tid * 4);
    __builtin_amdgcn_global_load_async_to_lds_b128(g, l, 0, 0);
  }
#else
  if (tid < 96) {
    float2 v = ((const float2*)gsrc)[tid];
    ((float2*)lbuf)[tid] = v;
  }
#endif
}

// ---------------------------------------------------------------------------
// Kernel 1: per-gaussian preprocessing.
// coeff8[i][0..7] = quadratic-form coefficients in WMMA K-order:
//   power(p) = c0*pu^2 + c1*pu*pv + c2*pv^2 + c3*pu + c4*pv + c5
// opacity has sigmoid + (z > NEAR) validity folded in; key = depth r.
// ---------------------------------------------------------------------------
__global__ void k_pre(const float* __restrict__ pos, const float* __restrict__ rgb,
                      const float* __restrict__ opa, const float* __restrict__ quat,
                      const float* __restrict__ scale, const float* __restrict__ rot,
                      const float* __restrict__ tran,
                      float* __restrict__ key, float* __restrict__ coeff8,
                      float* __restrict__ rgbu, float* __restrict__ opau, int n) {
  int i = blockIdx.x * blockDim.x + threadIdx.x;
  if (i >= n) return;

  float R[9];
#pragma unroll
  for (int j = 0; j < 9; ++j) R[j] = rot[j];

  float px = pos[3 * i + 0], py = pos[3 * i + 1], pz = pos[3 * i + 2];
  float cx = R[0] * px + R[1] * py + R[2] * pz + tran[0];
  float cy = R[3] * px + R[4] * py + R[5] * pz + tran[1];
  float cz = R[6] * px + R[7] * py + R[8] * pz + tran[2];
  float r = sqrtf(cx * cx + cy * cy + cz * cz);
  float invz = 1.0f / cz;
  float mx = cx * invz, my = cy * invz;

  float J00 = invz, J02 = -cx * invz * invz;
  float J11 = invz, J12 = -cy * invz * invz;

  float qw = quat[4 * i + 0], qx = quat[4 * i + 1], qy = quat[4 * i + 2], qz = quat[4 * i + 3];
  float qn = sqrtf(qw * qw + qx * qx + qy * qy + qz * qz) + 1e-12f;
  qw /= qn; qx /= qn; qy /= qn; qz /= qn;
  float Rm[9];
  Rm[0] = 1.0f - 2.0f * (qy * qy + qz * qz);
  Rm[1] = 2.0f * (qx * qy - qw * qz);
  Rm[2] = 2.0f * (qx * qz + qw * qy);
  Rm[3] = 2.0f * (qx * qy + qw * qz);
  Rm[4] = 1.0f - 2.0f * (qx * qx + qz * qz);
  Rm[5] = 2.0f * (qy * qz - qw * qx);
  Rm[6] = 2.0f * (qx * qz - qw * qy);
  Rm[7] = 2.0f * (qy * qz + qw * qx);
  Rm[8] = 1.0f - 2.0f * (qx * qx + qy * qy);

  float s0 = fabsf(scale[3 * i + 0]) + 1e-4f;
  float s1 = fabsf(scale[3 * i + 1]) + 1e-4f;
  float s2 = fabsf(scale[3 * i + 2]) + 1e-4f;

  float RS[9];
#pragma unroll
  for (int a = 0; a < 3; ++a) {
    RS[3 * a + 0] = Rm[3 * a + 0] * s0;
    RS[3 * a + 1] = Rm[3 * a + 1] * s1;
    RS[3 * a + 2] = Rm[3 * a + 2] * s2;
  }
  float C00 = RS[0] * RS[0] + RS[1] * RS[1] + RS[2] * RS[2];
  float C01 = RS[0] * RS[3] + RS[1] * RS[4] + RS[2] * RS[5];
  float C02 = RS[0] * RS[6] + RS[1] * RS[7] + RS[2] * RS[8];
  float C11 = RS[3] * RS[3] + RS[4] * RS[4] + RS[5] * RS[5];
  float C12 = RS[3] * RS[6] + RS[4] * RS[7] + RS[5] * RS[8];
  float C22 = RS[6] * RS[6] + RS[7] * RS[7] + RS[8] * RS[8];

  float JW0[3], JW1[3];
#pragma unroll
  for (int k = 0; k < 3; ++k) {
    JW0[k] = J00 * R[0 + k] + J02 * R[6 + k];
    JW1[k] = J11 * R[3 + k] + J12 * R[6 + k];
  }
  float M00 = JW0[0] * C00 + JW0[1] * C01 + JW0[2] * C02;
  float M01 = JW0[0] * C01 + JW0[1] * C11 + JW0[2] * C12;
  float M02 = JW0[0] * C02 + JW0[1] * C12 + JW0[2] * C22;
  float M10 = JW1[0] * C00 + JW1[1] * C01 + JW1[2] * C02;
  float M11 = JW1[0] * C01 + JW1[1] * C11 + JW1[2] * C12;
  float M12 = JW1[0] * C02 + JW1[1] * C12 + JW1[2] * C22;

  float c00 = M00 * JW0[0] + M01 * JW0[1] + M02 * JW0[2];
  float c01 = M00 * JW1[0] + M01 * JW1[1] + M02 * JW1[2];
  float c11 = M10 * JW1[0] + M11 * JW1[1] + M12 * JW1[2];

  float a = c00 + 1e-8f;
  float d = c11 + 1e-8f;
  float b = c01;
  float det = a * d - b * b;
  float ia = d / det, ib = -b / det, id = a / det;

  float c0 = -0.5f * ia;
  float c1 = -ib;
  float c2 = -0.5f * id;
  float c3 = ia * mx + ib * my;
  float c4 = ib * mx + id * my;
  float c5 = -0.5f * (ia * mx * mx + 2.0f * ib * mx * my + id * my * my);

  float* co = coeff8 + (size_t)i * 8;
  co[0] = c0; co[1] = c1; co[2] = c2; co[3] = c3;
  co[4] = c4; co[5] = c5; co[6] = 0.0f; co[7] = 0.0f;

  rgbu[3 * i + 0] = sigmoidf(rgb[3 * i + 0]);
  rgbu[3 * i + 1] = sigmoidf(rgb[3 * i + 1]);
  rgbu[3 * i + 2] = sigmoidf(rgb[3 * i + 2]);
  opau[i] = sigmoidf(opa[i]) * (cz > 0.3f ? 1.0f : 0.0f);
  key[i] = r;
}

// ---------------------------------------------------------------------------
// Kernel 2: single-workgroup bitonic sort by depth (32KB LDS), then gather
// into CHUNK-BLOCKED layout for async staging:
//   gdata[chunk][0..127]   = coeff8 of the 16 gaussians (slot*8 + k)
//   gdata[chunk][128..143] = opacity
//   gdata[chunk][144..191] = rgb (slot*3 + c)
// Pad slots -> zero (alpha = 0, contributes nothing).
// ---------------------------------------------------------------------------
__global__ __launch_bounds__(1024) void k_sort_gather(
    const float* __restrict__ key, const float* __restrict__ coeff8,
    const float* __restrict__ rgbu, const float* __restrict__ opau,
    float* __restrict__ gdata, int n) {
  __shared__ float sk[NPAD];
  __shared__ int si[NPAD];
  const int t = threadIdx.x;

  for (int e = t; e < NPAD; e += 1024) {
    sk[e] = (e < n) ? key[e] : 3.4e38f;
    si[e] = e;
  }
  __syncthreads();

  for (int k = 2; k <= NPAD; k <<= 1) {
    for (int j = k >> 1; j > 0; j >>= 1) {
      for (int e = t; e < NPAD; e += 1024) {
        int x = e ^ j;
        if (x > e) {
          bool up = ((e & k) == 0);
          float ka = sk[e], kb = sk[x];
          if ((ka > kb) == up) {
            sk[e] = kb; sk[x] = ka;
            int ti = si[e]; si[e] = si[x]; si[x] = ti;
          }
        }
      }
      __syncthreads();
    }
  }

  for (int e = t; e < NPAD; e += 1024) {
    int o = si[e];
    int ch = e >> 4, sl = e & 15;
    float* dg = gdata + (size_t)ch * CHF;
    if (o < n) {
      const float* so = coeff8 + (size_t)o * 8;
#pragma unroll
      for (int j = 0; j < 8; ++j) dg[sl * 8 + j] = so[j];
      dg[128 + sl] = opau[o];
      dg[144 + sl * 3 + 0] = rgbu[3 * o + 0];
      dg[144 + sl * 3 + 1] = rgbu[3 * o + 1];
      dg[144 + sl * 3 + 2] = rgbu[3 * o + 2];
    } else {
#pragma unroll
      for (int j = 0; j < 8; ++j) dg[sl * 8 + j] = 0.0f;
      dg[128 + sl] = 0.0f;
      dg[144 + sl * 3 + 0] = 0.0f;
      dg[144 + sl * 3 + 1] = 0.0f;
      dg[144 + sl * 3 + 2] = 0.0f;
    }
  }
}

// ---------------------------------------------------------------------------
// Kernel 3: WMMA-tiled splatting with double-buffered async-to-LDS staging.
// Block = 128 threads (4 waves x 16 pixels). Per 16-gaussian chunk:
//   - chunk c+1 staged to LDS via GLOBAL_LOAD_ASYNC_TO_LDS_B128 (ASYNCcnt)
//   - two V_WMMA_F32_16X16X4_F32 produce the 16x16 power tile
//     (M = gaussian, N = pixel); per-lane composite over 8 gaussians,
//     half-wave merge via shfl_xor(16)
//   - block-uniform early-out fused into __syncthreads_or
// ---------------------------------------------------------------------------
__global__ __launch_bounds__(128) void k_render(const float* __restrict__ gdata,
                                                float* __restrict__ out) {
  __shared__ __align__(16) float lds[2][CHF];

  const int tid = threadIdx.x;
  const int lane = tid & 31;
  const int wave = tid >> 5;
  const int h = lane >> 4;     // 0 = front 8 gaussians of chunk, 1 = back 8
  const int q = lane & 15;     // pixel column within the wave's 16-pixel tile
  const int pix = blockIdx.x * 64 + wave * 16 + q;
  const int u = pix % WW, v = pix / WW;
  const float pu = ((float)u + 0.5f - 0.5f * (float)WW) * (1.0f / 96.0f);
  const float pv = ((float)v + 0.5f - 0.5f * (float)HH) * (1.0f / 96.0f);

  float T = 1.0f, cr = 0.0f, cg = 0.0f, cb = 0.0f;

#if HAVE_WMMA_F32X4
  // B fragments (4x16 f32): VGPR0 = K(0|2), VGPR1 = K(1|3) per half-wave.
  v2f bA, bB;
  bA.x = h ? pv * pv : pu * pu;   // K2 : K0
  bA.y = h ? pu      : pu * pv;   // K3 : K1
  bB.x = h ? 0.0f    : pv;        // K6 : K4
  bB.y = h ? 0.0f    : 1.0f;      // K7 : K5
#else
  v16h bf = {};
  if (h == 0) {
    bf[0] = (_Float16)(pu * pu);
    bf[1] = (_Float16)(pu * pv);
    bf[2] = (_Float16)(pv * pv);
    bf[3] = (_Float16)pu;
    bf[4] = (_Float16)pv;
    bf[5] = (_Float16)1.0f;
  }
#endif

  // Prologue: stage chunk 0, wait, make visible to all waves.
  stage_chunk(gdata, lds[0], tid);
  wait_async();
  __syncthreads();

  for (int c = 0; c < NC; ++c) {
    if (c + 1 < NC) stage_chunk(gdata + (size_t)(c + 1) * CHF, lds[(c + 1) & 1], tid);
    if (c + 8 < NC) __builtin_prefetch(gdata + (size_t)(c + 8) * CHF, 0, 1);

    const float* buf = lds[c & 1];

    v8f acc = {};
#if HAVE_WMMA_F32X4
    // A fragments (16x4 f32): lane<16 -> M=lane, K0/K1 ; lane>=16 -> M=lane-16, K2/K3.
    const float* gp = buf + q * 8 + h * 2;
    v2f aA; aA.x = gp[0]; aA.y = gp[1];
    v2f aB; aB.x = gp[4]; aB.y = gp[5];
    acc = __builtin_amdgcn_wmma_f32_16x16x4_f32(false, aA, false, bA, (short)0, acc, false, false);
    acc = __builtin_amdgcn_wmma_f32_16x16x4_f32(false, aB, false, bB, (short)0, acc, false, false);
#else
    v16h af = {};
    if (h == 0) {
      const float* gp = buf + q * 8;
#pragma unroll
      for (int e = 0; e < 8; ++e) af[e] = (_Float16)gp[e];
    }
    acc = __builtin_amdgcn_wmma_f32_16x16x32_f16(false, af, false, bf, (short)0, acc, false, false);
#endif

    // Per-lane sequential composite over its 8 gaussians (M = c*16 + 8h + k).
    const float* op = buf + 128 + h * 8;
    const float* rp = buf + 144 + h * 24;
    float segT = 1.0f, sr = 0.0f, sg = 0.0f, sb = 0.0f;
#pragma unroll
    for (int k = 0; k < 8; ++k) {
      float al = op[k] * __expf(acc[k]);
      al = fminf(al, 0.99f);
      float wgt = segT * al;
      sr += wgt * rp[3 * k + 0];
      sg += wgt * rp[3 * k + 1];
      sb += wgt * rp[3 * k + 2];
      segT *= (1.0f - al);
    }

    // Merge half-wave segments in depth order (h=0 first), branchless.
    float oT = __shfl_xor(segT, 16, 32);
    float orr = __shfl_xor(sr, 16, 32);
    float og = __shfl_xor(sg, 16, 32);
    float ob = __shfl_xor(sb, 16, 32);
    float fT = h ? oT : segT;
    float f_r = h ? orr : sr, f_g = h ? og : sg, f_b = h ? ob : sb;
    float s_r = h ? sr : orr, s_g = h ? sg : og, s_b = h ? sb : ob;
    float ccr = f_r + fT * s_r;
    float ccg = f_g + fT * s_g;
    float ccb = f_b + fT * s_b;
    float cT = segT * oT;

    cr += T * ccr; cg += T * ccg; cb += T * ccb;
    T *= cT;

    // Fused: async-complete wait + block barrier + block-uniform early-out.
    wait_async();
    int alive = __syncthreads_or(T > 1e-7f ? 1 : 0);
    if (!alive) break;
  }

  if (h == 0) {
    float* o = out + (size_t)pix * 3;
    o[0] = cr; o[1] = cg; o[2] = cb;
  }
}

// ---------------------------------------------------------------------------
extern "C" void kernel_launch(void* const* d_in, const int* in_sizes, int n_in,
                              void* d_out, int out_size, void* d_ws, size_t ws_size,
                              hipStream_t stream) {
  const float* pos = (const float*)d_in[0];
  const float* rgb = (const float*)d_in[1];
  const float* opa = (const float*)d_in[2];
  const float* quat = (const float*)d_in[3];
  const float* scale = (const float*)d_in[4];
  const float* rot = (const float*)d_in[5];
  const float* tran = (const float*)d_in[6];
  int n = in_sizes[2];
  if (n > NPAD) n = NPAD;

  float* f = (float*)d_ws;
  float* key = f;                       // 4096
  float* coeff8 = f + NPAD;             // 4096*8
  float* rgbu = f + NPAD * 9;           // 4096*3
  float* opau = f + NPAD * 12;          // 4096
  float* gdata = f + NPAD * 13;         // 256 chunks * 192 floats = 4096*12

  k_pre<<<(n + 255) / 256, 256, 0, stream>>>(pos, rgb, opa, quat, scale, rot, tran,
                                             key, coeff8, rgbu, opau, n);
  k_sort_gather<<<1, 1024, 0, stream>>>(key, coeff8, rgbu, opau, gdata, n);
  k_render<<<(HH * WW) / 64, 128, 0, stream>>>(gdata, (float*)d_out);
}
